// SGAT_37538014167318
// MI455X (gfx1250) — matrix operand order repaired and checked by reading, not compiled, not run on hardware
//
#include <hip/hip_runtime.h>
#include <hip/hip_bf16.h>
#include <math.h>

typedef __attribute__((ext_vector_type(16))) _Float16 v16h;
typedef __attribute__((ext_vector_type(8)))  float    v8f;
typedef __attribute__((ext_vector_type(4)))  float    f32x4;

#define N_NODES 4096
#define CIN     128
#define COUT    64
#define BATCH   2

// ---------------------------------------------------------------------------
// Kernel 1: h = relu(relu(x@W1+b1)@W2+b2);  u = h@Wa[:C]; v = h@Wa[C:]
// One block = 16 nodes, 256 threads. W1/W2/biases staged in LDS (~58KB).
// ---------------------------------------------------------------------------
__global__ __launch_bounds__(256, 1) void k1_mlp(
    const float* __restrict__ x,  const float* __restrict__ W1,
    const float* __restrict__ b1, const float* __restrict__ W2,
    const float* __restrict__ b2, const float* __restrict__ Wa,
    float* __restrict__ h, float* __restrict__ u, float* __restrict__ v)
{
    __shared__ float W1s[CIN * COUT];
    __shared__ float W2s[COUT * COUT];
    __shared__ float h1s[16 * COUT];
    __shared__ float h2s[16 * COUT];
    __shared__ float sWa[2 * COUT];
    __shared__ float sb1[COUT];
    __shared__ float sb2[COUT];

    const int t  = threadIdx.x;
    const int b  = blockIdx.x >> 8;            // 256 tiles per batch
    const int i0 = (blockIdx.x & 255) << 4;    // 16 rows per tile

    // vectorized LDS fill of the weights
    {
        const f32x4* W1v = (const f32x4*)W1;
        f32x4*       W1sv = (f32x4*)W1s;
        for (int i = t; i < (CIN * COUT) / 4; i += 256) W1sv[i] = W1v[i];
        const f32x4* W2v = (const f32x4*)W2;
        f32x4*       W2sv = (f32x4*)W2s;
        for (int i = t; i < (COUT * COUT) / 4; i += 256) W2sv[i] = W2v[i];
    }
    if (t < 2 * COUT) sWa[t] = Wa[t];
    if (t < COUT) { sb1[t] = b1[t]; sb2[t] = b2[t]; }
    __syncthreads();

    const int c  = t & 63;
    const int r0 = t >> 6;                     // 0..3

    // layer 1: 16x64 outputs, 4 per thread
    #pragma unroll
    for (int q = 0; q < 4; ++q) {
        const int r = r0 + q * 4;
        const f32x4* xr = (const f32x4*)(x + ((size_t)(b * N_NODES) + i0 + r) * CIN);
        float acc = sb1[c];
        for (int k4 = 0; k4 < CIN / 4; ++k4) {
            const f32x4 xv = xr[k4];
            acc = fmaf(xv.x, W1s[(k4 * 4 + 0) * COUT + c], acc);
            acc = fmaf(xv.y, W1s[(k4 * 4 + 1) * COUT + c], acc);
            acc = fmaf(xv.z, W1s[(k4 * 4 + 2) * COUT + c], acc);
            acc = fmaf(xv.w, W1s[(k4 * 4 + 3) * COUT + c], acc);
        }
        h1s[r * COUT + c] = fmaxf(acc, 0.0f);
    }
    __syncthreads();

    // layer 2
    #pragma unroll
    for (int q = 0; q < 4; ++q) {
        const int r = r0 + q * 4;
        float acc = sb2[c];
        for (int k = 0; k < COUT; ++k) acc = fmaf(h1s[r * COUT + k], W2s[k * COUT + c], acc);
        acc = fmaxf(acc, 0.0f);
        h2s[r * COUT + c] = acc;
        h[((size_t)(b * N_NODES) + i0 + r) * COUT + c] = acc;
    }
    __syncthreads();

    // u (threads 0..15) and v (threads 16..31)
    if (t < 32) {
        const int r = t & 15;
        const float* wv = (t < 16) ? sWa : (sWa + COUT);
        float acc = 0.0f;
        for (int k = 0; k < COUT; ++k) acc = fmaf(h2s[r * COUT + k], wv[k], acc);
        if (t < 16) u[b * N_NODES + i0 + r] = acc;
        else        v[b * N_NODES + i0 + r] = acc;
    }
}

// ---------------------------------------------------------------------------
// Kernel 2: repack h (f32) into f16 WMMA B-fragments (32x16 tiles, KxN).
// Lane layout used here (32x16 B, contiguous k, same as A): lanes 0-15 hold K=0..15, lanes 16-31 hold
// K=16..31; N = lane%16; VGPR p packs {K=2p (lo), K=2p+1 (hi)}.
// frag index = ((b*128 + kChunk)*4 + nTile)*32 + lane, 8 dwords each.
// ---------------------------------------------------------------------------
__global__ void k2_pack(const float* __restrict__ h, unsigned int* __restrict__ hB)
{
    const int idx  = blockIdx.x * 256 + threadIdx.x;   // 0..32767
    const int lane = idx & 31;
    const int nt   = (idx >> 5) & 3;
    const int kc   = (idx >> 7) & 127;
    const int b    = idx >> 14;

    const int n  = nt * 16 + (lane & 15);
    const int kb = kc * 32 + ((lane < 16) ? 0 : 16);
    const float* hb = h + (size_t)b * N_NODES * COUT;
    unsigned int* out = hB + (size_t)idx * 8;

    #pragma unroll
    for (int p = 0; p < 8; ++p) {
        _Float16 lo = (_Float16)hb[(size_t)(kb + 2 * p)     * COUT + n];
        _Float16 hi = (_Float16)hb[(size_t)(kb + 2 * p + 1) * COUT + n];
        unsigned short lob = __builtin_bit_cast(unsigned short, lo);
        unsigned short hib = __builtin_bit_cast(unsigned short, hi);
        out[p] = (unsigned int)lob | ((unsigned int)hib << 16);
    }
}

// ---------------------------------------------------------------------------
// Kernel 3: attention tile. 16 rows/block, 256 threads (8 waves).
// LDS (dynamic, ~305KB): e_tile[16][4096] f32, red[8][16][64], part[16][256],
// segs[256], inv[16], u[16].
// adj is streamed with non-temporal loads, a written with non-temporal stores
// (read-once / write-once: keep them out of L2 so hB/h/v stay resident).
// ---------------------------------------------------------------------------
__global__ __launch_bounds__(256, 1) void k3_attn(
    const float* __restrict__ adj, const float* __restrict__ u_g,
    const float* __restrict__ v_g, const float* __restrict__ ba,
    const unsigned int* __restrict__ hB,
    float* __restrict__ hout, float* __restrict__ a_out)
{
    extern __shared__ float smem[];
    float* e_tile = smem;                   // 65536 floats
    float* red    = e_tile + 16 * N_NODES;  // 8192
    float* part   = red + 8192;             // 4096
    float* segs   = part + 4096;            // 256
    float* inv_s  = segs + 256;             // 16
    float* u_s    = inv_s + 16;             // 16

    const int tid  = threadIdx.x;
    const int lane = tid & 31;
    const int w    = tid >> 5;
    const int b    = blockIdx.x >> 8;
    const int i0   = (blockIdx.x & 255) << 4;

    if (tid < 16) u_s[tid] = u_g[b * N_NODES + i0 + tid];
    __syncthreads();
    const float bav = ba[0];
    const f32x4* vr4 = (const f32x4*)(v_g + b * N_NODES);

    // ---- phase 1: e_ij = sigmoid(u_i+v_j+ba)*adj, partial row sums ----
    // 128-bit vectorized: 1024 float4 per row, 4 iters per thread per row.
    for (int r = 0; r < 16; ++r) {
        const float  ur = u_s[r] + bav;
        const f32x4* adjr4 = (const f32x4*)(adj + ((size_t)(b * N_NODES) + i0 + r) * N_NODES);
        f32x4* er4 = (f32x4*)(e_tile + r * N_NODES);
        float local = 0.0f;
        for (int j4 = tid; j4 < N_NODES / 4; j4 += 256) {
            const f32x4 av = __builtin_nontemporal_load(adjr4 + j4);  // TH=NT
            const f32x4 vv = vr4[j4];
            f32x4 e;
            e.x = av.x / (1.0f + __expf(-(ur + vv.x)));
            e.y = av.y / (1.0f + __expf(-(ur + vv.y)));
            e.z = av.z / (1.0f + __expf(-(ur + vv.z)));
            e.w = av.w / (1.0f + __expf(-(ur + vv.w)));
            er4[j4] = e;                                              // b128 DS store
            local += (e.x + e.y) + (e.z + e.w);
        }
        part[r * 256 + tid] = local;
    }
    __syncthreads();

    // ---- deterministic row-sum reduction ----
    {
        const int r = tid >> 4, s = tid & 15;
        float acc = 0.0f;
        #pragma unroll
        for (int m = 0; m < 16; ++m) acc += part[r * 256 + s * 16 + m];
        segs[r * 16 + s] = acc;
    }
    __syncthreads();
    if (tid < 16) {
        float acc = 0.0f;
        #pragma unroll
        for (int s = 0; s < 16; ++s) acc += segs[tid * 16 + s];
        inv_s[tid] = 1.0f / (acc + 1e-10f);
    }
    __syncthreads();

    // ---- phase 2: write normalized a (128-bit non-temporal stores) ----
    for (int r = 0; r < 16; ++r) {
        const float ir = inv_s[r];
        f32x4* ar4 = (f32x4*)(a_out + ((size_t)(b * N_NODES) + i0 + r) * N_NODES);
        const f32x4* er4 = (const f32x4*)(e_tile + r * N_NODES);
        for (int j4 = tid; j4 < N_NODES / 4; j4 += 256) {
            f32x4 e = er4[j4];
            e.x *= ir; e.y *= ir; e.z *= ir; e.w *= ir;
            __builtin_nontemporal_store(e, ar4 + j4);                 // TH=NT
        }
    }

    // ---- phase 3: h_out tile = a @ h via v_wmma_f32_16x16x32_f16 ----
    // wave w owns K in [w*512, w*512+512); 16 chunks of 32.
    // A k order used here (16x32, contiguous, same as hB): lanes<16: M=lane, K=0..15;
    // lanes>=16: M=lane-16, K=16..31.
    const int   rA    = lane & 15;
    const float irA   = inv_s[rA];
    const int   koffA = (lane < 16) ? 0 : 16;
    v8f acc[4] = {};
    const unsigned int* hBb = hB + (size_t)b * (128 * 4 * 32 * 8);

    for (int ch = 0; ch < 16; ++ch) {
        const int k0 = w * 512 + ch * 32;
        const f32x4* g1 = (const f32x4*)(e_tile + rA * N_NODES + k0 + koffA);
        const f32x4* g2 = (const f32x4*)(e_tile + rA * N_NODES + k0 + 8 + koffA);
        const f32x4 f0 = g1[0], f1 = g1[1], f2 = g2[0], f3 = g2[1];
        v16h af;
        af[0]  = (_Float16)(f0.x * irA); af[1]  = (_Float16)(f0.y * irA);
        af[2]  = (_Float16)(f0.z * irA); af[3]  = (_Float16)(f0.w * irA);
        af[4]  = (_Float16)(f1.x * irA); af[5]  = (_Float16)(f1.y * irA);
        af[6]  = (_Float16)(f1.z * irA); af[7]  = (_Float16)(f1.w * irA);
        af[8]  = (_Float16)(f2.x * irA); af[9]  = (_Float16)(f2.y * irA);
        af[10] = (_Float16)(f2.z * irA); af[11] = (_Float16)(f2.w * irA);
        af[12] = (_Float16)(f3.x * irA); af[13] = (_Float16)(f3.y * irA);
        af[14] = (_Float16)(f3.z * irA); af[15] = (_Float16)(f3.w * irA);

        const int kc = k0 >> 5;
        #pragma unroll
        for (int t4 = 0; t4 < 4; ++t4) {
            const uint4* bp = (const uint4*)(hBb + ((size_t)(kc * 4 + t4) * 32 + lane) * 8);
            union { uint4 u4[2]; v16h v; } bu;
            bu.u4[0] = bp[0]; bu.u4[1] = bp[1];
            acc[t4] = __builtin_amdgcn_wmma_f32_16x16x32_f16(
                false, af, false, bu.v, (short)0, acc[t4], false, false);
        }
    }

    // ---- cross-wave reduction of 8 partial 16x64 tiles ----
    // C layout: VGPR q, lane L -> M = q + (L<16?0:8), N = L%16.
    #pragma unroll
    for (int t4 = 0; t4 < 4; ++t4) {
        #pragma unroll
        for (int q = 0; q < 8; ++q) {
            const int m = q + ((lane < 16) ? 0 : 8);
            const int n = t4 * 16 + (lane & 15);
            red[(w * 16 + m) * 64 + n] = acc[t4][q];
        }
    }
    __syncthreads();
    for (int e = tid; e < 16 * 64; e += 256) {
        float s = 0.0f;
        #pragma unroll
        for (int ww = 0; ww < 8; ++ww) s += red[ww * 1024 + e];
        const int m = e >> 6, cc = e & 63;
        hout[((size_t)(b * N_NODES) + i0 + m) * 64 + cc] = s;
    }
}

// ---------------------------------------------------------------------------
extern "C" void kernel_launch(void* const* d_in, const int* in_sizes, int n_in,
                              void* d_out, int out_size, void* d_ws, size_t ws_size,
                              hipStream_t stream)
{
    const float* x   = (const float*)d_in[0];
    const float* adj = (const float*)d_in[1];
    const float* W1  = (const float*)d_in[2];
    const float* b1  = (const float*)d_in[3];
    const float* W2  = (const float*)d_in[4];
    const float* b2  = (const float*)d_in[5];
    const float* Wa  = (const float*)d_in[6];
    const float* ba  = (const float*)d_in[7];

    float* hout  = (float*)d_out;                                   // [B,N,64]
    float* a_out = hout + (size_t)BATCH * N_NODES * COUT;           // [B,N,N]

    char* ws = (char*)d_ws;
    float*        h  = (float*)ws;                                  // 2 MB
    float*        u  = (float*)(ws + 2097152);                      // 32 KB
    float*        v  = (float*)(ws + 2097152 + 32768);              // 32 KB
    unsigned int* hB = (unsigned int*)(ws + 2097152 + 65536);       // 1 MB

    k1_mlp<<<BATCH * (N_NODES / 16), 256, 0, stream>>>(x, W1, b1, W2, b2, Wa, h, u, v);
    k2_pack<<<(BATCH * 128 * 4 * 32) / 256, 256, 0, stream>>>(h, hB);

    const size_t shmem = (size_t)(16 * N_NODES + 8192 + 4096 + 256 + 16 + 16) * sizeof(float);
    k3_attn<<<BATCH * (N_NODES / 16), 256, shmem, stream>>>(adj, u, v, ba, hB, hout, a_out);
}
